// GraphConvLayer_61366492725742
// MI455X (gfx1250) — matrix operand order repaired and checked by reading, not compiled, not run on hardware
//
#include <hip/hip_runtime.h>

// ---------------------------------------------------------------------------
// CDNA5 / gfx1250 GCN layer.  B=16, N=1024, HID=768, HEADS=12, HD=64
//   self_loop = (adj@out + out) @ W^T  ==  (adj + I) @ (out @ W^T)
//   => project first (K=w, N=64), then adj@P (K=1024, N=64).
//   Matmuls: v_wmma_f32_16x16x32_bf16 (f32 accum).  Hot kernel (k_agg) stages
//   the shared B operand (P^T) in LDS via the Tensor Data Mover (TDM), one
//   2D descriptor per 64KB K-slab, synced with s_wait_tensorcnt.
// ---------------------------------------------------------------------------

typedef __bf16 bf16_t;
typedef __attribute__((ext_vector_type(16))) __bf16 v16bf;
typedef __attribute__((ext_vector_type(8)))  __bf16 v8bf;
typedef __attribute__((ext_vector_type(8)))  float  v8f;
typedef __attribute__((ext_vector_type(4)))  unsigned int u32x4;
typedef __attribute__((ext_vector_type(8)))  int i32x8;
typedef __attribute__((ext_vector_type(4)))  int i32x4;

#define NB   16
#define NN   1024
#define HID  768
#define NH   12
#define HD   64
#define LDO  1536   // padded width of concat buffer (max 1472)
#define KSLAB 512   // K-chunk of P^T staged in LDS (64 x 512 bf16 = 64 KB)

// ---- WMMA fragment loaders (bf16 16x16x32, wave32 ISA layouts) -------------
// A 16x32: lanes 0-15 -> M=lane, K 0-7 (v0-3) and 16-23 (v4-7);
//          lanes 16-31 -> same M, K 8-15 and 24-31.  => k = e + (e&8) + 8*h
__device__ __forceinline__ v16bf load_fragA(const bf16_t* __restrict__ A,
                                            int lda, int row, int kk, int h) {
    const v8bf lo = *(const v8bf*)(A + (size_t)row * lda + kk + 8 * h);
    const v8bf hi = *(const v8bf*)(A + (size_t)row * lda + kk + 16 + 8 * h);
    v16bf a;
#pragma unroll
    for (int e = 0; e < 8; ++e) { a[e] = lo[e]; a[e + 8] = hi[e]; }
    return a;
}
// B 32x16: same K order as A: lanes 0-15 K 0-7,16-23; lanes 16-31 K 8-15,24-31.
// Bt is the *transposed* B (column-contiguous): Bt[n][k].  => k = e + (e&8) + 8*h
__device__ __forceinline__ v16bf load_fragB(const bf16_t* Bt,
                                            int ldb, int col, int kk, int h) {
    const v8bf b0 = *(const v8bf*)(Bt + (size_t)col * ldb + kk + 8 * h);
    const v8bf b1 = *(const v8bf*)(Bt + (size_t)col * ldb + kk + 16 + 8 * h);
    v16bf b;
#pragma unroll
    for (int e = 0; e < 8; ++e) { b[e] = b0[e]; b[e + 8] = b1[e]; }
    return b;
}

// ---- Tensor Data Mover: 2D tile (rows x tile_k bf16, row stride stride_k) --
#if __has_builtin(__builtin_amdgcn_tensor_load_to_lds)
#define HAVE_TDM 1
__device__ __forceinline__ void tdm_load_2d_bf16(unsigned lds_off,
                                                 const void* gaddr,
                                                 int tile_k, int rows,
                                                 int tensor_k, int stride_k) {
    unsigned long long ga = (unsigned long long)gaddr;
    u32x4 g0 = {};
    g0[0] = 1u;                                   // count=1 (valid), no gather
    g0[1] = lds_off;                              // lds_addr (bytes)
    g0[2] = (unsigned)ga;                         // global_addr[31:0]
    g0[3] = (unsigned)((ga >> 32) & 0x1FFFFFFu)   // global_addr[56:32]
          | (2u << 30);                           // type=2 ("image")
    i32x8 g1 = {};
    g1[0] = 1 << 16;                              // data_size=1 (2 bytes)
    g1[1] = (tensor_k & 0xFFFF) << 16;            // tensor_dim0[15:0]
    g1[2] = ((tensor_k >> 16) & 0xFFFF)           // tensor_dim0[31:16]
          | ((rows & 0xFFFF) << 16);              // tensor_dim1[15:0]
    g1[3] = ((rows >> 16) & 0xFFFF)               // tensor_dim1[31:16]
          | ((tile_k & 0xFFFF) << 16);            // tile_dim0
    g1[4] = rows & 0xFFFF;                        // tile_dim1 (tile_dim2=0)
    g1[5] = stride_k;                             // tensor_dim0_stride[31:0]
    g1[6] = 0;                                    // stride hi / dim1_stride lo
    g1[7] = 0;
    i32x4 g2 = {};
    i32x4 g3 = {};
#if defined(__clang_major__) && __clang_major__ >= 23
    i32x8 g4 = {};
    __builtin_amdgcn_tensor_load_to_lds(g0, g1, g2, g3, g4, 0);
#else
    __builtin_amdgcn_tensor_load_to_lds(g0, g1, g2, g3, 0);
#endif
}
#endif

// ---- Conversion / reduction kernels ---------------------------------------

__global__ __launch_bounds__(256) void k_adj_denom(const float* __restrict__ adj,
                                                   bf16_t* __restrict__ adjBF,
                                                   float* __restrict__ rdenom) {
    const int row = blockIdx.x;                  // b*N + n
    const float* src = adj + (size_t)row * NN;
    bf16_t* dst = adjBF + (size_t)row * NN;
    float s = 0.f;
    for (int c = threadIdx.x; c < NN; c += 256) {
        float v = src[c];
        s += v;
        dst[c] = (bf16_t)v;
    }
    __shared__ float red[256];
    red[threadIdx.x] = s;
    __syncthreads();
    for (int off = 128; off > 0; off >>= 1) {
        if (threadIdx.x < off) red[threadIdx.x] += red[threadIdx.x + off];
        __syncthreads();
    }
    if (threadIdx.x == 0) rdenom[row] = 1.0f / (red[0] + 1.0f);
}

__global__ __launch_bounds__(256) void k_feat2bf(const float* __restrict__ feat,
                                                 bf16_t* __restrict__ outBF,
                                                 int total) {
    int i = blockIdx.x * 256 + threadIdx.x;
    if (i >= total) return;
    int c = i % HID;
    int bn = i / HID;
    outBF[(size_t)bn * LDO + c] = (bf16_t)feat[i];
}

__global__ __launch_bounds__(256) void k_f2bf(const float* __restrict__ src,
                                              bf16_t* __restrict__ dst, int n) {
    int i = blockIdx.x * 256 + threadIdx.x;
    if (i < n) dst[i] = (bf16_t)src[i];
}

__global__ __launch_bounds__(256) void k_resid(const float* __restrict__ norms,
                                               const float* __restrict__ feat,
                                               bf16_t* __restrict__ dst, int total) {
    int i = blockIdx.x * 256 + threadIdx.x;
    if (i < total) dst[i] = (bf16_t)(norms[i] + feat[i]);
}

// ---- P = output @ W^T  (M=1024 rows per batch, K=w, N=64) ------------------
__global__ __launch_bounds__(32) void k_proj(const bf16_t* __restrict__ outBF,
                                             const bf16_t* __restrict__ WBF,
                                             float* __restrict__ PF32,
                                             bf16_t* __restrict__ PBFt,
                                             int w) {
    const int b = blockIdx.y;
    const int m0 = blockIdx.x * 16;
    const int lane = threadIdx.x & 31;
    const int r = lane & 15, h = lane >> 4;
    const bf16_t* A = outBF + (size_t)b * NN * LDO;

    v8f acc[4] = {};
    for (int kk = 0; kk < w; kk += 32) {
        v16bf a = load_fragA(A, LDO, m0 + r, kk, h);
#pragma unroll
        for (int c = 0; c < 4; ++c) {
            v16bf bf = load_fragB(WBF, w, c * 16 + r, kk, h);
            acc[c] = __builtin_amdgcn_wmma_f32_16x16x32_bf16(
                false, a, false, bf, (short)0, acc[c], false, false);
        }
    }
    float* Pb = PF32 + (size_t)b * NN * HD;
    bf16_t* Pt = PBFt + (size_t)b * HD * NN;
#pragma unroll
    for (int c = 0; c < 4; ++c) {
#pragma unroll
        for (int j = 0; j < 8; ++j) {
            int m = m0 + j + 8 * h;
            int n = c * 16 + r;
            float v = acc[c][j];
            Pb[m * HD + n] = v;
            Pt[n * NN + m] = (bf16_t)v;   // transposed (column-contiguous)
        }
    }
}

// ---- norm = lrelu(((adj+I)@P + 2b)/denom); append to concat buffer ---------
// 8 waves / block, 16 rows x 64 cols per wave (128 rows per block).
// B operand (P^T slab) staged in 64KB LDS via TDM, shared by all 8 waves.
__global__ __launch_bounds__(256) void k_agg(const bf16_t* __restrict__ adjBF,
                                             const bf16_t* __restrict__ PBFt,
                                             const float* __restrict__ PF32,
                                             const float* __restrict__ bias,
                                             const float* __restrict__ rdenom,
                                             float* __restrict__ normsF32,
                                             bf16_t* __restrict__ outBF,
                                             int head) {
    __shared__ __align__(16) bf16_t Bsh[HD * KSLAB];   // 64 x 512 bf16 = 64 KB

    const int b = blockIdx.y;
    const int wv = threadIdx.x >> 5;
    const int m0 = blockIdx.x * 128 + wv * 16;
    const int lane = threadIdx.x & 31;
    const int r = lane & 15, h = lane >> 4;
    const bf16_t* A  = adjBF + (size_t)b * NN * NN;
    const bf16_t* Bt = PBFt + (size_t)b * HD * NN;

    v8f acc[4] = {};
    for (int s = 0; s < NN / KSLAB; ++s) {
        const int k0 = s * KSLAB;
#if defined(HAVE_TDM)
        if (wv == 0) {                        // wave-uniform guard (TDM ignores EXEC)
            tdm_load_2d_bf16((unsigned)(size_t)&Bsh[0], Bt + k0,
                             KSLAB, HD, NN, NN);
            __builtin_amdgcn_s_wait_tensorcnt(0);
        }
#else
        for (int idx = threadIdx.x; idx < HD * KSLAB / 8; idx += 256) {
            int row = idx / (KSLAB / 8);
            int chk = idx % (KSLAB / 8);
            *(v8bf*)(Bsh + row * KSLAB + chk * 8) =
                *(const v8bf*)(Bt + (size_t)row * NN + k0 + chk * 8);
        }
#endif
        __syncthreads();
        for (int kk = 0; kk < KSLAB; kk += 32) {
            if (kk + 32 < KSLAB)             // keep the L2-resident adj stream hot
                __builtin_prefetch(A + (size_t)(m0 + r) * NN + k0 + kk + 32 + 8 * h, 0, 3);
            v16bf a = load_fragA(A, NN, m0 + r, k0 + kk, h);
#pragma unroll
            for (int c = 0; c < 4; ++c) {
                v16bf bf = load_fragB(Bsh, KSLAB, c * 16 + r, kk, h);
                acc[c] = __builtin_amdgcn_wmma_f32_16x16x32_bf16(
                    false, a, false, bf, (short)0, acc[c], false, false);
            }
        }
        __syncthreads();                      // LDS reuse for next slab
    }

    const float* Pb = PF32 + (size_t)b * NN * HD;
    float* nrm = normsF32 + (size_t)b * NN * HID;
    bf16_t* ob = outBF + (size_t)b * NN * LDO;
#pragma unroll
    for (int c = 0; c < 4; ++c) {
#pragma unroll
        for (int j = 0; j < 8; ++j) {
            int m = m0 + j + 8 * h;
            int n = c * 16 + r;
            float v = acc[c][j] + Pb[m * HD + n] + 2.0f * bias[n];
            float t = v * rdenom[b * NN + m];
            t = (t > 0.f) ? t : 0.01f * t;            // leaky_relu(0.01)
            nrm[m * HID + head * HD + n] = t;
            ob[(size_t)m * LDO + HID + head * HD + n] = (bf16_t)t;
        }
    }
}

// ---- out = residual @ Wt^T + bt  ------------------------------------------
__global__ __launch_bounds__(32) void k_final(const bf16_t* __restrict__ residBF,
                                              const bf16_t* __restrict__ WtBF,
                                              const float* __restrict__ bt,
                                              float* __restrict__ out) {
    const int b = blockIdx.z;
    const int m0 = blockIdx.x * 16;
    const int n0 = blockIdx.y * 64;
    const int lane = threadIdx.x & 31;
    const int r = lane & 15, h = lane >> 4;
    const bf16_t* A = residBF + (size_t)b * NN * HID;

    v8f acc[4] = {};
    for (int kk = 0; kk < HID; kk += 32) {
        v16bf a = load_fragA(A, HID, m0 + r, kk, h);
#pragma unroll
        for (int c = 0; c < 4; ++c) {
            v16bf bf = load_fragB(WtBF, HID, n0 + c * 16 + r, kk, h);
            acc[c] = __builtin_amdgcn_wmma_f32_16x16x32_bf16(
                false, a, false, bf, (short)0, acc[c], false, false);
        }
    }
    float* ob = out + (size_t)b * NN * HID;
#pragma unroll
    for (int c = 0; c < 4; ++c) {
#pragma unroll
        for (int j = 0; j < 8; ++j) {
            int m = m0 + j + 8 * h;
            int n = n0 + c * 16 + r;
            ob[m * HID + n] = acc[c][j] + bt[n];
        }
    }
}

// ---------------------------------------------------------------------------
extern "C" void kernel_launch(void* const* d_in, const int* in_sizes, int n_in,
                              void* d_out, int out_size, void* d_ws, size_t ws_size,
                              hipStream_t stream) {
    (void)in_sizes; (void)out_size; (void)ws_size;

    const float* adj  = (const float*)d_in[0];
    const float* feat = (const float*)d_in[1];

    int w[NH];
    for (int i = 0; i < NH; ++i) w[i] = HID + HD * i;

    const float* Wh[NH];
    const float* bh[NH];
    const float* Wt;
    const float* btv;
    if (n_in >= 28) {                 // tuples flattened to separate inputs
        for (int i = 0; i < NH; ++i) { Wh[i] = (const float*)d_in[2 + i];
                                       bh[i] = (const float*)d_in[14 + i]; }
        Wt  = (const float*)d_in[26];
        btv = (const float*)d_in[27];
    } else {                          // tuples passed concatenated
        const float* wcat = (const float*)d_in[2];
        const float* bcat = (const float*)d_in[3];
        size_t o = 0;
        for (int i = 0; i < NH; ++i) { Wh[i] = wcat + o; o += (size_t)HD * w[i];
                                       bh[i] = bcat + (size_t)HD * i; }
        Wt  = (const float*)d_in[4];
        btv = (const float*)d_in[5];
    }

    // ---- workspace carve-up (~170 MB) ----
    char* ws = (char*)d_ws;
    size_t off = 0;
    auto carve = [&](size_t bytes) -> void* {
        void* p = ws + off;
        off = (off + bytes + 255) & ~(size_t)255;
        return p;
    };
    bf16_t* adjBF   = (bf16_t*)carve((size_t)NB * NN * NN * 2);     // 32 MB
    bf16_t* outBF   = (bf16_t*)carve((size_t)NB * NN * LDO * 2);    // 48 MB
    float*  normsF  = (float*) carve((size_t)NB * NN * HID * 4);    // 48 MB
    float*  PF32    = (float*) carve((size_t)NB * NN * HD * 4);     //  4 MB
    bf16_t* PBFt    = (bf16_t*)carve((size_t)NB * HD * NN * 2);     //  2 MB
    float*  rdenom  = (float*) carve((size_t)NB * NN * 4);
    bf16_t* residBF = (bf16_t*)carve((size_t)NB * NN * HID * 2);    // 24 MB
    bf16_t* WtBF    = (bf16_t*)carve((size_t)HID * HID * 2);
    bf16_t* WBF     = (bf16_t*)carve((size_t)HD * 13440 * 2);       // all heads

    size_t woff[NH];
    { size_t o = 0; for (int i = 0; i < NH; ++i) { woff[i] = o; o += (size_t)HD * w[i]; } }

    // ---- 1. conversions + denom ----
    k_adj_denom<<<NB * NN, 256, 0, stream>>>(adj, adjBF, rdenom);
    {
        int total = NB * NN * HID;
        k_feat2bf<<<(total + 255) / 256, 256, 0, stream>>>(feat, outBF, total);
    }
    for (int i = 0; i < NH; ++i) {
        int n = HD * w[i];
        k_f2bf<<<(n + 255) / 256, 256, 0, stream>>>(Wh[i], WBF + woff[i], n);
    }
    {
        int n = HID * HID;
        k_f2bf<<<(n + 255) / 256, 256, 0, stream>>>(Wt, WtBF, n);
    }

    // ---- 2. sequential heads: project then (adj+I)@P ----
    for (int i = 0; i < NH; ++i) {
        dim3 gp(NN / 16, NB);
        k_proj<<<gp, 32, 0, stream>>>(outBF, WBF + woff[i], PF32, PBFt, w[i]);
        dim3 ga(NN / 128, NB);
        k_agg<<<ga, 256, 0, stream>>>(adjBF, PBFt, PF32, bh[i], rdenom,
                                      normsF, outBF, i);
    }

    // ---- 3. residual + output projection ----
    {
        int total = NB * NN * HID;
        k_resid<<<(total + 255) / 256, 256, 0, stream>>>(normsF, feat, residBF, total);
    }
    {
        dim3 g(NN / 16, HID / 64, NB);
        k_final<<<g, 32, 0, stream>>>(residBF, WtBF, btv, (float*)d_out);
    }
}
